// GreedyMatcher_10892037062884
// MI455X (gfx1250) — matrix operand order repaired and checked by reading, not compiled, not run on hardware
//
#include <hip/hip_runtime.h>
#include <hip/hip_bf16.h>
#include <stdint.h>

// ---------------------------------------------------------------------------
// PDHG LP solver for the GreedyMatcher reference, CDNA5 (gfx1250) wave32.
//   m = 128 constraints, n = 2048 structures, batch 256, 100 iterations.
//   GEMMs via v_wmma_f32_16x16x32_bf16 (S is 0/1 -> exact in bf16).
//   One workgroup (16 waves) per 16-row batch tile; all state on-chip.
// ---------------------------------------------------------------------------

typedef __attribute__((ext_vector_type(16))) __bf16 v16bf;
typedef __attribute__((ext_vector_type(8)))  __bf16 v8bf;
typedef __attribute__((ext_vector_type(8)))  float  v8f;
typedef unsigned short u16;

// global (addrspace 1) views so laundered pointers still lower to
// global_load_b128 (LOADcnt only) instead of flat_load_b128 (LOADcnt+DScnt)
typedef __attribute__((address_space(1))) const u16   gu16;
typedef __attribute__((address_space(1))) const v16bf gv16bf;

#define M_DIM   128     // constraints (N_HOS*N_TYPES)
#define N_DIM   2048    // structures
#define BATCH   256
#define N_ITERS 100

#define XS 2056         // xbar LDS row stride in bf16 elems (4112B rows -> bank-conflict free)
#define YS 136          // y/ysum LDS row stride (fp32 / bf16 elems)

#define SM_XBAR 0                       // u16 [16][XS]   = 65792 B
#define SM_YBF  (16*XS*2)               // u16 [16][YS]   =  4352 B
#define SM_YOLD (SM_YBF + 16*YS*2)      // f32 [16][YS]   =  8704 B
#define SM_YSUM (SM_YOLD + 16*YS*4)     // f32 [16][YS]   =  8704 B
#define SM_BYTES (SM_YSUM + 16*YS*4)    // 87552 B dynamic LDS

__device__ __forceinline__ u16 f2bf(float f) {
  unsigned int u = __float_as_uint(f);
  unsigned int r = u + 0x7FFFu + ((u >> 16) & 1u);   // round-to-nearest-even
  return (u16)(r >> 16);
}

// ---- convert S (fp32 [128,2048]) to bf16 row-major and bf16 transpose ------
__global__ void cvt_s_kernel(const float* __restrict__ S,
                             u16* __restrict__ Sbf, u16* __restrict__ Stbf) {
  int idx = blockIdx.x * blockDim.x + threadIdx.x;
  if (idx >= M_DIM * N_DIM) return;
  int r = idx / N_DIM, c = idx % N_DIM;
  u16 h = f2bf(S[idx]);
  Sbf[idx] = h;                 // [m][n] row-major
  Stbf[c * M_DIM + r] = h;      // [n][m] row-major (= S^T)
}

// ---- spectral norm of S via 30-step power iteration; writes tau=0.9/L ------
__global__ void specnorm_kernel(const float* __restrict__ S, float* __restrict__ tau_out) {
  __shared__ float v[N_DIM];
  __shared__ float u[M_DIM];
  __shared__ float red[256];
  int tid = threadIdx.x;
  float inv = rsqrtf((float)N_DIM);
  for (int i = tid; i < N_DIM; i += 256) v[i] = inv;
  __syncthreads();

  for (int it = 0; it < 30; ++it) {
    if (tid < M_DIM) {                       // u = S v
      float s = 0.f;
      const float* row = S + tid * N_DIM;
      for (int k = 0; k < N_DIM; ++k) s += row[k] * v[k];
      u[tid] = s;
    }
    __syncthreads();
    red[tid] = (tid < M_DIM) ? u[tid] * u[tid] : 0.f;
    __syncthreads();
    for (int st = 128; st > 0; st >>= 1) { if (tid < st) red[tid] += red[tid + st]; __syncthreads(); }
    float nu = sqrtf(red[0]) + 1e-12f;
    __syncthreads();
    if (tid < M_DIM) u[tid] /= nu;
    __syncthreads();
    for (int c = tid; c < N_DIM; c += 256) { // v = S^T u
      float s = 0.f;
      for (int r = 0; r < M_DIM; ++r) s += S[r * N_DIM + c] * u[r];
      v[c] = s;
    }
    __syncthreads();
    float q = 0.f;
    for (int c = tid; c < N_DIM; c += 256) q += v[c] * v[c];
    red[tid] = q;
    __syncthreads();
    for (int st = 128; st > 0; st >>= 1) { if (tid < st) red[tid] += red[tid + st]; __syncthreads(); }
    float nv = sqrtf(red[0]) + 1e-12f;
    __syncthreads();
    for (int c = tid; c < N_DIM; c += 256) v[c] /= nv;
    __syncthreads();
  }
  if (tid < M_DIM) {                         // L = ||S v||
    float s = 0.f;
    const float* row = S + tid * N_DIM;
    for (int k = 0; k < N_DIM; ++k) s += row[k] * v[k];
    u[tid] = s;
  }
  __syncthreads();
  red[tid] = (tid < M_DIM) ? u[tid] * u[tid] : 0.f;
  __syncthreads();
  for (int st = 128; st > 0; st >>= 1) { if (tid < st) red[tid] += red[tid + st]; __syncthreads(); }
  if (tid == 0) tau_out[0] = 0.9f / (sqrtf(red[0]) + 1e-12f);
}

// ---- main PDHG kernel: 16 blocks x 512 threads (16 waves), 16 rows/WG ------
__global__ void __launch_bounds__(512)
pdhg_kernel(const float* __restrict__ Xg,      // [256,128] bids (b)
            const u16*   __restrict__ Sbf,     // [128,2048] bf16
            const u16*   __restrict__ Stbf,    // [2048,128] bf16
            const float* __restrict__ Wg,      // [2048]
            const float* __restrict__ tau_g,
            float*       __restrict__ out)     // [256,2048]
{
  extern __shared__ char smem[];
  u16*   xbarbf = (u16*)(smem + SM_XBAR);
  u16*   ybf    = (u16*)(smem + SM_YBF);
  float* yold   = (float*)(smem + SM_YOLD);
  float* ysum   = (float*)(smem + SM_YSUM);

  const int tid  = threadIdx.x;
  const int wid  = tid >> 5;          // wave 0..15 -> owns n-slice [wid*128, +128)
  const int lane = tid & 31;
  const int b0   = blockIdx.x * 16;   // batch tile base row

  const float tau   = tau_g[0];
  const float sigma = tau;

  // WMMA lane decomposition (wave32):
  const int arow = lane & 15;             // A-frag: matrix row per lane
  const int asel = (lane >> 4) * 16;      // A-frag: K-half select
  const int bcol = lane & 15;             // B-frag: matrix column per lane
  const int bsel = (lane >> 4) * 16;      // B-frag: K-half select
  const int mrow = (lane >> 4) * 8;       // C/D: row base per lane half

  const int kwbase = wid * 128;           // this wave's K-slice (GEMM1) / n-slice (GEMM2)

  // x accumulators: 8 tiles of 16x16 (8 VGPRs each) per wave
  v8f xacc[8];
  v8f vzero = {};
#pragma unroll
  for (int nt = 0; nt < 8; ++nt) xacc[nt] = vzero;

  // loop-invariant global values cached in registers (no global loads in loop
  // except the S/S^T fragment stream)
  float wv[8];
#pragma unroll
  for (int nt = 0; nt < 8; ++nt) wv[nt] = Wg[kwbase + nt * 16 + bcol];

  const int yrow = tid >> 5;              // 0..15 (y-update ownership)
  const int ycb  = (tid & 31) * 4;        // 4 cols per thread
  float bidv[4];
#pragma unroll
  for (int e = 0; e < 4; ++e) bidv[e] = Xg[(b0 + yrow) * M_DIM + ycb + e];

  // init LDS state (x0 = y0 = xbar0 = 0)
  for (int i = tid; i < 16 * XS; i += 512) xbarbf[i] = 0;
  for (int i = tid; i < 16 * YS; i += 512) { yold[i] = 0.f; ysum[i] = 0.f; ybf[i] = 0; }
  __syncthreads();

  // Laundered base addresses: redefined by empty asm each iteration so the
  // compiler cannot hoist (and spill) the B-fragment loads across iterations.
  // Re-cast to addrspace(1) so they still lower to global_load_b128.
  uintptr_t sp  = (uintptr_t)Sbf;
  uintptr_t stp = (uintptr_t)Stbf;

  for (int it = 0; it < N_ITERS; ++it) {
    asm volatile("" : "+s"(sp), "+s"(stp));
    gu16* Sp  = (gu16*)sp;
    gu16* Stp = (gu16*)stp;

    // ---------------- GEMM1: ysum += xbar @ S^T (partial over K-slice) ------
    v16bf Af[4];
#pragma unroll
    for (int kf = 0; kf < 4; ++kf) {
      int k0 = kwbase + kf * 32;
      const u16* base = xbarbf + arow * XS + k0 + asel;
      v8bf lo = *(const v8bf*)(base);
      v8bf hi = *(const v8bf*)(base + 8);
      Af[kf] = __builtin_shufflevector(lo, hi, 0,1,2,3,4,5,6,7,8,9,10,11,12,13,14,15);
    }
#pragma unroll
    for (int t = 0; t < 8; ++t) {         // y column tiles (128 = 8 x 16)
      v8f acc = vzero;
#pragma unroll
      for (int kf = 0; kf < 4; ++kf) {
        int k0 = kwbase + kf * 32;
        v16bf B = *(gv16bf*)(Sp + (16 * t + bcol) * N_DIM + k0 + bsel);
        acc = __builtin_amdgcn_wmma_f32_16x16x32_bf16(false, Af[kf], false, B,
                                                      (short)0, acc, false, false);
      }
      int col = 16 * t + bcol;
#pragma unroll
      for (int e = 0; e < 8; ++e)
        atomicAdd(&ysum[(mrow + e) * YS + col], acc[e]);   // ds_add_f32 reduce
    }
    __syncthreads();

    // ---------------- dual update: y = relu(y + sigma*(ysum - b)) -----------
#pragma unroll
    for (int e = 0; e < 4; ++e) {
      int c = ycb + e;
      float s  = ysum[yrow * YS + c];
      float yn = fmaxf(yold[yrow * YS + c] + sigma * (s - bidv[e]), 0.f);
      yold[yrow * YS + c] = yn;
      ybf [yrow * YS + c] = f2bf(yn);
      ysum[yrow * YS + c] = 0.f;          // re-zero for next iteration
    }
    __syncthreads();

    // ---------------- GEMM2: delta = y @ S ; primal update + extrapolation --
    v16bf Ay[4];
#pragma unroll
    for (int kf = 0; kf < 4; ++kf) {
      int k0 = kf * 32;
      const u16* base = ybf + arow * YS + k0 + asel;
      v8bf lo = *(const v8bf*)(base);
      v8bf hi = *(const v8bf*)(base + 8);
      Ay[kf] = __builtin_shufflevector(lo, hi, 0,1,2,3,4,5,6,7,8,9,10,11,12,13,14,15);
    }
#pragma unroll
    for (int nt = 0; nt < 8; ++nt) {
      int n0 = kwbase + nt * 16;
      v8f acc = vzero;
#pragma unroll
      for (int kf = 0; kf < 4; ++kf) {
        v16bf B = *(gv16bf*)(Stp + (n0 + bcol) * M_DIM + kf * 32 + bsel);
        acc = __builtin_amdgcn_wmma_f32_16x16x32_bf16(false, Ay[kf], false, B,
                                                      (short)0, acc, false, false);
      }
      int col = n0 + bcol;
#pragma unroll
      for (int e = 0; e < 8; ++e) {
        float xo = xacc[nt][e];
        float xn = fmaxf(xo + tau * (wv[nt] - acc[e]), 0.f);
        float xb = 2.f * xn - xo;                    // extrapolation
        xacc[nt][e] = xn;
        xbarbf[(mrow + e) * XS + col] = f2bf(xb);
      }
    }
    __syncthreads();
  }

  // ---------------- store x ------------------------------------------------
#pragma unroll
  for (int nt = 0; nt < 8; ++nt) {
    int col = kwbase + nt * 16 + bcol;
#pragma unroll
    for (int e = 0; e < 8; ++e)
      out[(b0 + mrow + e) * N_DIM + col] = xacc[nt][e];
  }
}

// ---------------------------------------------------------------------------
extern "C" void kernel_launch(void* const* d_in, const int* in_sizes, int n_in,
                              void* d_out, int out_size, void* d_ws, size_t ws_size,
                              hipStream_t stream) {
  const float* X = (const float*)d_in[0];   // [256,8,16] == [256,128]
  const float* S = (const float*)d_in[1];   // [128,2048]
  const float* W = (const float*)d_in[2];   // [2048]
  (void)in_sizes; (void)n_in; (void)out_size; (void)ws_size;

  u16*   Sbf   = (u16*)d_ws;                                  // 512 KB
  u16*   Stbf  = (u16*)((char*)d_ws + M_DIM * N_DIM * 2);     // 512 KB
  float* tau_w = (float*)((char*)d_ws + 2 * M_DIM * N_DIM * 2);

  cvt_s_kernel<<<(M_DIM * N_DIM + 255) / 256, 256, 0, stream>>>(S, Sbf, Stbf);
  specnorm_kernel<<<1, 256, 0, stream>>>(S, tau_w);
  pdhg_kernel<<<BATCH / 16, 512, SM_BYTES, stream>>>(X, Sbf, Stbf, W, tau_w,
                                                     (float*)d_out);
}
